// Attention_8899172237379
// MI455X (gfx1250) — matrix operand order repaired and checked by reading, not compiled, not run on hardware
//
#include <hip/hip_runtime.h>

typedef unsigned short u16;
typedef __attribute__((ext_vector_type(16))) __bf16 bf16x16;
typedef __attribute__((ext_vector_type(8)))  float  f32x8;
typedef __attribute__((ext_vector_type(4)))  unsigned int u32x4;
typedef __attribute__((ext_vector_type(8)))  unsigned int u32x8;

// xor-shuffle within the 32-lane wave via LDS-swizzle unit (no LDS memory touched).
#define LANE_XOR_F(v, m) \
  __int_as_float(__builtin_amdgcn_ds_swizzle(__float_as_int(v), (((m) << 10) | 0x1f)))

static __device__ __forceinline__ u16 f32_to_bf16(float f) {
  unsigned int u = __float_as_uint(f);
  u += 0x7fffu + ((u >> 16) & 1u);   // round-to-nearest-even
  return (u16)(u >> 16);
}

union V16BF { uint4 u[2]; bf16x16 v; };

static __device__ __forceinline__ bf16x16 ld_frag2(const u16* p0, const u16* p1) {
  V16BF r;
  r.u[0] = *reinterpret_cast<const uint4*>(p0);
  r.u[1] = *reinterpret_cast<const uint4*>(p1);
  return r.v;
}

static __device__ __forceinline__ f32x8 wmma_bf16(bf16x16 a, bf16x16 b, f32x8 c) {
  return __builtin_amdgcn_wmma_f32_16x16x32_bf16(false, a, false, b, (short)0, c,
                                                 false, false);
}

// ---- CDNA5 async / tensor data movement -----------------------------------

static __device__ __forceinline__ void async_copy_b128(const void* g, void* l) {
  const unsigned long long ga = (unsigned long long)(uintptr_t)g;
  const unsigned la = (unsigned)(uintptr_t)l;  // LDS generic ptr: low 32 = LDS addr
  asm volatile("global_load_async_to_lds_b128 %0, %1, off"
               :: "v"(la), "v"(ga) : "memory");
}

template <int N>
static __device__ __forceinline__ void wait_async() {
#if __has_builtin(__builtin_amdgcn_s_wait_asynccnt)
  __builtin_amdgcn_s_wait_asynccnt(N);
#else
  asm volatile("s_wait_asynccnt %0" :: "i"(N) : "memory");
#endif
}

static __device__ __forceinline__ void wait_tensor0() {
#if __has_builtin(__builtin_amdgcn_s_wait_tensorcnt)
  __builtin_amdgcn_s_wait_tensorcnt(0);
#else
  asm volatile("s_wait_tensorcnt 0x0" ::: "memory");
#endif
}

// TDM: DMA a 2D tile (tile_d0 x tile_d1, 16-bit elems) from a strided tensor into
// LDS.  D# groups per CDNA5 ISA 8.3/8.4; 2-group form (2D, VADDR2/3 = NULL).
static __device__ __forceinline__ void tdm_load_2d_bf16(
    const void* gaddr, unsigned lds_addr, unsigned tensor_d0, unsigned tensor_d1,
    unsigned d0_stride, unsigned tile_d0, unsigned tile_d1) {
  const unsigned long long ga = (unsigned long long)(uintptr_t)gaddr;
  u32x4 g0;
  g0[0] = 1u;                                            // count=1, user D#
  g0[1] = lds_addr;
  g0[2] = (unsigned)ga;
  g0[3] = (unsigned)((ga >> 32) & 0x01FFFFFFu) | (2u << 30);  // addr[56:32], type=2
  u32x8 g1;
  g1[0] = 1u << 16;                                      // data_size = 2 bytes
  g1[1] = (tensor_d0 & 0xFFFFu) << 16;
  g1[2] = ((tensor_d0 >> 16) & 0xFFFFu) | ((tensor_d1 & 0xFFFFu) << 16);
  g1[3] = ((tensor_d1 >> 16) & 0xFFFFu) | ((tile_d0 & 0xFFFFu) << 16);
  g1[4] = tile_d1 & 0xFFFFu;
  g1[5] = d0_stride;
  g1[6] = 0u;
  g1[7] = 0u;
  asm volatile("tensor_load_to_lds %0, %1" :: "s"(g0), "s"(g1) : "memory");
}

// ---------------------------------------------------------------------------

static __device__ __forceinline__ float half16_max(float v) {
  v = fmaxf(v, LANE_XOR_F(v, 1));
  v = fmaxf(v, LANE_XOR_F(v, 2));
  v = fmaxf(v, LANE_XOR_F(v, 4));
  v = fmaxf(v, LANE_XOR_F(v, 8));
  return v;
}
static __device__ __forceinline__ float half16_sum(float v) {
  v += LANE_XOR_F(v, 1);
  v += LANE_XOR_F(v, 2);
  v += LANE_XOR_F(v, 4);
  v += LANE_XOR_F(v, 8);
  return v;
}

__global__ void f32_to_bf16_kernel(const float* __restrict__ in,
                                   u16* __restrict__ out, int n) {
  for (int i = blockIdx.x * blockDim.x + threadIdx.x; i < n;
       i += gridDim.x * blockDim.x)
    out[i] = f32_to_bf16(in[i]);
}

// C[M,N] = A[M,K] * Bm[N,K]^T + bias[N].  bf16 row-major inputs (K contiguous).
// 256 threads = 8 waves; tile 128(M) x 64(N); K in 64-chunks, double-buffered
// LDS fed by ASYNCcnt-tracked global->LDS DMA (6 B128 copies / thread / tile).
template <bool STORE_F32>
__global__ __launch_bounds__(256) void gemm_tn_bf16(
    const u16* __restrict__ A, const u16* __restrict__ Bm,
    const float* __restrict__ bias, void* __restrict__ Cout,
    int M, int N, int K) {
  __shared__ __align__(16) u16 As[2][128 * 64];
  __shared__ __align__(16) u16 Bs[2][64 * 64];
  const int t  = threadIdx.x;
  const int ln = t & 31, w = t >> 5;
  const int hs = ln >> 4, lm = ln & 15;
  const int m0 = blockIdx.x * 128, n0 = blockIdx.y * 64;
  f32x8 acc[4] = {};

  auto stage = [&](int kc, int buf) {
#pragma unroll
    for (int i = 0; i < 4; ++i) {           // A tile 128x64: 1024 B128 DMAs
      const int l = t + i * 256;
      const int row = l >> 3, c8 = (l & 7) * 8;
      async_copy_b128(&A[(size_t)(m0 + row) * K + kc + c8],
                      &As[buf][row * 64 + c8]);
    }
#pragma unroll
    for (int i = 0; i < 2; ++i) {           // B tile 64x64: 512 B128 DMAs
      const int l = t + i * 256;
      const int row = l >> 3, c8 = (l & 7) * 8;
      async_copy_b128(&Bm[(size_t)(n0 + row) * K + kc + c8],
                      &Bs[buf][row * 64 + c8]);
    }
  };

  const int T = K / 64;
  stage(0, 0);
  for (int it = 0; it < T; ++it) {
    const int cur = it & 1;
    if (it + 1 < T) {
      stage((it + 1) * 64, cur ^ 1);        // next tile in flight behind WMMA
      wait_async<6>();                      // only waits for *this* tile's DMAs
    } else {
      wait_async<0>();
    }
    __syncthreads();

    const u16* ar = &As[cur][(w * 16 + lm) * 64];
#pragma unroll
    for (int c = 0; c < 2; ++c) {           // two K=32 sub-chunks
      const bf16x16 a = ld_frag2(ar + c * 32 + 16 * hs, ar + c * 32 + 16 * hs + 8);
#pragma unroll
      for (int nf = 0; nf < 4; ++nf) {
        const u16* br = &Bs[cur][(nf * 16 + lm) * 64 + c * 32 + 16 * hs];
        acc[nf] = wmma_bf16(a, ld_frag2(br, br + 8), acc[nf]);
      }
    }
    __syncthreads();                        // tile fully consumed before reuse
  }

#pragma unroll
  for (int nf = 0; nf < 4; ++nf) {
    const int col = n0 + nf * 16 + lm;
    const float bv = bias[col];
#pragma unroll
    for (int r = 0; r < 8; ++r) {           // C layout: VGPR r -> row r + 8*hs
      const int row = m0 + w * 16 + r + 8 * hs;
      const float v = acc[nf][r] + bv;
      if (STORE_F32)
        reinterpret_cast<float*>(Cout)[(size_t)row * N + col] = v;
      else
        reinterpret_cast<u16*>(Cout)[(size_t)row * N + col] = f32_to_bf16(v);
    }
  }
}

// Flash attention: grid (S/64, B*H), 128 threads = 4 waves; each wave owns a
// 16-query tile; 32-key blocks, online softmax, all WMMA bf16.  K tiles are
// TDM descriptors double-buffered one block ahead; V tiles load early into
// registers and scatter transposed after compute.  One barrier per block.
__global__ __launch_bounds__(128) void flash_attn_bf16(
    const u16* __restrict__ qkv, u16* __restrict__ ctx) {
  constexpr int S = 2048, E = 1024, E3 = 3072, D = 64;
  __shared__ __align__(16) u16 Ks[2][32 * 64];   // [key][d]  (TDM dest)
  __shared__ __align__(16) u16 Vt[2][64 * 32];   // [d][key]  (transposed)
  __shared__ __align__(16) u16 Pb[4 * 16 * 32];  // per-wave P re-layout scratch
  const int t  = threadIdx.x, ln = t & 31, w = t >> 5;
  const int hs = ln >> 4, lm = ln & 15;
  const int b  = blockIdx.y >> 4, h = blockIdx.y & 15;
  const int qbase = blockIdx.x * 64 + w * 16;

  const u16* qrow = qkv + (size_t)(b * S + qbase + lm) * E3 + h * D;
  const bf16x16 aq0 = ld_frag2(qrow + 16 * hs, qrow + 16 * hs + 8);
  const bf16x16 aq1 = ld_frag2(qrow + 32 + 16 * hs, qrow + 40 + 16 * hs);

  auto stage_k = [&](int jb, int buf) {     // one TDM descriptor per K tile
    if (w == 0)
      tdm_load_2d_bf16(qkv + (size_t)(b * S + jb * 32) * E3 + E + h * D,
                       (unsigned)(uintptr_t)&Ks[buf][0],
                       /*tensor_d0=*/D, /*tensor_d1=*/S,
                       /*d0_stride=*/E3, /*tile_d0=*/D, /*tile_d1=*/32);
  };
  auto load_v = [&](int jb, uint4* vr) {    // V block -> registers (early issue)
#pragma unroll
    for (int i = 0; i < 2; ++i) {
      const int l = t + i * 128;
      const int kr = l >> 3, c8 = (l & 7) * 8;
      vr[i] = *reinterpret_cast<const uint4*>(
          &qkv[(size_t)(b * S + jb * 32 + kr) * E3 + h * D + 2 * E + c8]);
    }
  };
  auto store_v = [&](const uint4* vr, int buf) {  // scatter transposed into LDS
#pragma unroll
    for (int i = 0; i < 2; ++i) {
      const int l = t + i * 128;
      const int kr = l >> 3, c8 = (l & 7) * 8;
      const u16* vs = reinterpret_cast<const u16*>(&vr[i]);
#pragma unroll
      for (int j = 0; j < 8; ++j) Vt[buf][(c8 + j) * 32 + kr] = vs[j];
    }
  };

  f32x8 o[4] = {};
  float mrow[8], lrow[8];
#pragma unroll
  for (int r = 0; r < 8; ++r) { mrow[r] = -3.0e38f; lrow[r] = 0.f; }

  // prologue: stage block 0
  uint4 vreg[2];
  stage_k(0, 0);
  load_v(0, vreg);
  store_v(vreg, 0);
  if (w == 0) wait_tensor0();
  __syncthreads();

  for (int jb = 0; jb < S / 32; ++jb) {
    const int cur = jb & 1;
    const bool more = (jb + 1) < (S / 32);
    if (more) {
      stage_k(jb + 1, cur ^ 1);             // TDM in flight behind compute
      load_v(jb + 1, vreg);                 // global loads issue early
    }

    // S = Q * K^T : two 16-key column groups over dims 0..31 / 32..63
    f32x8 s0 = {}, s1 = {};
    {
      const u16* k0 = &Ks[cur][lm * 64];
      s0 = wmma_bf16(aq0, ld_frag2(k0 + 16 * hs, k0 + 16 * hs + 8), s0);
      s0 = wmma_bf16(aq1, ld_frag2(k0 + 32 + 16 * hs, k0 + 40 + 16 * hs), s0);
      const u16* k1 = &Ks[cur][(16 + lm) * 64];
      s1 = wmma_bf16(aq0, ld_frag2(k1 + 16 * hs, k1 + 16 * hs + 8), s1);
      s1 = wmma_bf16(aq1, ld_frag2(k1 + 32 + 16 * hs, k1 + 40 + 16 * hs), s1);
    }

    // online softmax over the 32 keys (scale = 1/sqrt(64))
    u16* pw = &Pb[w * 512];
#pragma unroll
    for (int r = 0; r < 8; ++r) {
      const float v0 = s0[r] * 0.125f, v1 = s1[r] * 0.125f;
      const float mx   = half16_max(fmaxf(v0, v1));
      const float mnew = fmaxf(mrow[r], mx);
      const float p0 = __expf(v0 - mnew), p1 = __expf(v1 - mnew);
      const float ps = half16_sum(p0 + p1);
      const float alpha = __expf(mrow[r] - mnew);
      lrow[r] = lrow[r] * alpha + ps;
      mrow[r] = mnew;
      o[0][r] *= alpha; o[1][r] *= alpha; o[2][r] *= alpha; o[3][r] *= alpha;
      pw[(r + 8 * hs) * 32 + lm]      = f32_to_bf16(p0);
      pw[(r + 8 * hs) * 32 + 16 + lm] = f32_to_bf16(p1);
    }

    // reload P as an A fragment (16x32) — same-wave LDS RAW, dscnt handles it
    const u16* pr = &Pb[w * 512 + lm * 32];
  const bf16x16 ap = ld_frag2(pr + 16 * hs, pr + 16 * hs + 8);

    // O += P * V : 4 d-column groups; B frag contiguous thanks to Vt transpose
#pragma unroll
    for (int n4 = 0; n4 < 4; ++n4) {
      const u16* vr = &Vt[cur][(n4 * 16 + lm) * 32 + 16 * hs];
      o[n4] = wmma_bf16(ap, ld_frag2(vr, vr + 8), o[n4]);
    }

    if (more) {
      store_v(vreg, cur ^ 1);               // other buffer: no reader conflict
      if (w == 0) wait_tensor0();           // TDM had full compute to land
    }
    __syncthreads();                        // publish next block's tiles
  }

  // normalize and emit context (bf16) for the output projection
#pragma unroll
  for (int n4 = 0; n4 < 4; ++n4) {
#pragma unroll
    for (int r = 0; r < 8; ++r) {
      const int row = qbase + r + 8 * hs;
      const int col = h * D + n4 * 16 + lm;
      ctx[(size_t)(b * S + row) * E + col] = f32_to_bf16(o[n4][r] / lrow[r]);
    }
  }
}

extern "C" void kernel_launch(void* const* d_in, const int* in_sizes, int n_in,
                              void* d_out, int out_size, void* d_ws, size_t ws_size,
                              hipStream_t stream) {
  const float* x     = (const float*)d_in[0];
  const float* w_in  = (const float*)d_in[1];
  const float* b_in  = (const float*)d_in[2];
  const float* w_out = (const float*)d_in[3];
  const float* b_out = (const float*)d_in[4];

  constexpr int B = 2, S = 2048, E = 1024;
  constexpr int M = B * S;  // 4096 token rows

  u16* xb    = (u16*)d_ws;                       // [M, E]
  u16* winb  = xb    + (size_t)M * E;            // [3E, E]
  u16* woutb = winb  + (size_t)3 * E * E;        // [E, E]
  u16* qkv   = woutb + (size_t)E * E;            // [M, 3E]
  u16* ctx   = qkv   + (size_t)M * 3 * E;        // [M, E]

  f32_to_bf16_kernel<<<2048, 256, 0, stream>>>(x,     xb,    M * E);
  f32_to_bf16_kernel<<<2048, 256, 0, stream>>>(w_in,  winb,  3 * E * E);
  f32_to_bf16_kernel<<<1024, 256, 0, stream>>>(w_out, woutb, E * E);

  gemm_tn_bf16<false><<<dim3(M / 128, (3 * E) / 64), 256, 0, stream>>>(
      xb, winb, b_in, qkv, M, 3 * E, E);

  flash_attn_bf16<<<dim3(S / 64, B * 16), 128, 0, stream>>>(qkv, ctx);

  gemm_tn_bf16<true><<<dim3(M / 128, E / 64), 256, 0, stream>>>(
      ctx, woutb, b_out, d_out, M, E, E);
}
